// S4ClassEncoder_55267639165352
// MI455X (gfx1250) — matrix operand order repaired and checked by reading, not compiled, not run on hardware
//
#include <hip/hip_runtime.h>
#include <hip/hip_bf16.h>
#include <cstdint>
#include <cstddef>

// Problem constants (match reference)
#define BB   16
#define LL   8192
#define DD   128
#define NN   32
#define EE   256
#define NBK  6
#define NCLS 16

typedef __bf16 bf16_t;
typedef bf16_t v16bf __attribute__((ext_vector_type(16)));
typedef bf16_t v8bf  __attribute__((ext_vector_type(8)));
typedef float  v8f   __attribute__((ext_vector_type(8)));
typedef float  v4f   __attribute__((ext_vector_type(4)));

__device__ __forceinline__ bf16_t f2bf(float f) {
  uint32_t u = __float_as_uint(f);
  uint32_t r = (u + 0x7fffu + ((u >> 16) & 1u)) >> 16;  // round-to-nearest-even
  unsigned short s = (unsigned short)r;
  return __builtin_bit_cast(bf16_t, s);
}

__device__ __forceinline__ float gelu_f(float x) {
  const float c0 = 0.7978845608028654f;  // sqrt(2/pi), tanh-approx gelu (jax default)
  float x3 = x * x * x;
  return 0.5f * x * (1.0f + tanhf(c0 * (x + 0.044715f * x3)));
}
__device__ __forceinline__ float sigmoid_f(float x) {
  return 1.0f / (1.0f + __expf(-x));
}

// ---------------------------------------------------------------------------
// WMMA fragment helpers, exact gfx1250 16-bit VGPR layouts (ISA 7.12.2).
// A (16x32, M=pos, K): lanes 0-15 hold K = k0+0..7 & k0+16..23 for row M=lane;
//                      lanes 16-31 hold K = k0+8..15 & k0+24..31, row M=lane-16.
// B (32x16, K x N=col): lanes 0-15 hold K = k0+0..7 & k0+16..23 for col N=lane;
//                       lanes 16-31 hold K = k0+8..15 & k0+24..31, col N=lane-16.
// C/D (16x16 f32): lane<16 -> rows 0..7 of col N=lane; lane>=16 -> rows 8..15.
// ---------------------------------------------------------------------------
__device__ __forceinline__ v16bf load_a_frag(const bf16_t* rowp, int k0, int hi) {
  int ka = k0 + (hi ? 8 : 0);
  const v8bf lo = *(const v8bf*)(rowp + ka);
  const v8bf hh = *(const v8bf*)(rowp + ka + 16);
  v16bf a;
#pragma unroll
  for (int i = 0; i < 8; ++i) { a[i] = lo[i]; a[i + 8] = hh[i]; }
  return a;
}
__device__ __forceinline__ v16bf load_b_frag(const bf16_t* wrow, int k0, int hi) {
  int kb = k0 + (hi ? 8 : 0);
  const v8bf lo = *(const v8bf*)(wrow + kb);
  const v8bf hh = *(const v8bf*)(wrow + kb + 16);
  v16bf b;
#pragma unroll
  for (int i = 0; i < 8; ++i) { b[i] = lo[i]; b[i + 8] = hh[i]; }
  return b;
}
__device__ __forceinline__ v8f wmma_bf16(v16bf a, v16bf b, v8f c) {
  return __builtin_amdgcn_wmma_f32_16x16x32_bf16(false, a, false, b, (short)0, c,
                                                 false, false);
}

// ---------------------------------------------------------------------------
// One-time f32 -> bf16 weight conversion (all three weight tensors are the
// same element count: NBK*256*128).
// ---------------------------------------------------------------------------
__global__ void cvt_weights_kernel(const float* __restrict__ ow,
                                   const float* __restrict__ w1,
                                   const float* __restrict__ w2,
                                   bf16_t* __restrict__ owb,
                                   bf16_t* __restrict__ w1b,
                                   bf16_t* __restrict__ w2b, int n) {
  int i = blockIdx.x * blockDim.x + threadIdx.x;
  if (i >= n) return;
  owb[i] = f2bf(ow[i]);
  w1b[i] = f2bf(w1[i]);
  w2b[i] = f2bf(w2[i]);
}

// ---------------------------------------------------------------------------
// Depthwise-from-1ch conv: h[b,d,l] = conv_b[d] + sum_k w[d,0,k] * x[b,l+k-2]
// ---------------------------------------------------------------------------
__global__ __launch_bounds__(256) void conv_kernel(const float* __restrict__ x,
                                                   const float* __restrict__ cw,
                                                   const float* __restrict__ cb,
                                                   float* __restrict__ h) {
  size_t idx = (size_t)blockIdx.x * blockDim.x + threadIdx.x;
  int l = (int)(idx % LL);
  int rem = (int)(idx / LL);
  int d = rem % DD;
  int b = rem / DD;
  const float* xr = x + (size_t)b * LL;
  float acc = cb[d];
#pragma unroll
  for (int k = 0; k < 5; ++k) {
    int li = l + k - 2;
    if (li >= 0 && li < LL) acc += cw[d * 5 + k] * xr[li];
  }
  h[idx] = acc;
}

// ---------------------------------------------------------------------------
// Channel LayerNorm per (b,l). mode 0: write f32 (B,D,L) for the SSM scan.
// mode 1: write bf16 transposed (B,L,D) as the WMMA A-operand buffer.
// ---------------------------------------------------------------------------
__global__ __launch_bounds__(256) void ln_kernel(const float* __restrict__ h,
                                                 float* __restrict__ zf,
                                                 bf16_t* __restrict__ zb,
                                                 const float* __restrict__ w,
                                                 const float* __restrict__ bia,
                                                 int to_bf) {
  int tid = blockIdx.x * blockDim.x + threadIdx.x;  // over B*L
  int b = tid / LL, l = tid % LL;
  const float* col = h + (size_t)b * DD * LL + l;   // stride LL across channels
  float s = 0.f, s2 = 0.f;
  for (int d = 0; d < DD; ++d) {
    float v = col[(size_t)d * LL];
    s += v; s2 += v * v;
  }
  float mu = s * (1.0f / DD);
  float var = s2 * (1.0f / DD) - mu * mu;
  float rs = rsqrtf(var + 1e-5f);
  if (to_bf) {
    bf16_t* out = zb + ((size_t)b * LL + l) * DD;
    for (int d = 0; d < DD; ++d)
      out[d] = f2bf((col[(size_t)d * LL] - mu) * rs * w[d] + bia[d]);
  } else {
    float* out = zf + (size_t)b * DD * LL + l;
    for (int d = 0; d < DD; ++d)
      out[(size_t)d * LL] = (col[(size_t)d * LL] - mu) * rs * w[d] + bia[d];
  }
}

// ---------------------------------------------------------------------------
// S4 diagonal-SSM recurrence (exact equivalent of the FFT long-conv since the
// linear conv length 2L-1 <= nf). One wave32 per (b,d); lane n holds state n.
//   lambda = exp(dt*A);  Ctil = 2*(C_re+iC_im)*(exp(dtA)-1)/A
//   s = lambda*s + z[l];  y[l] = Re(sum_n Ctil_n * s_n)
// Then g = gelu(y + Dskip*z) written bf16 into the (B,L,D) GEMM-A buffer.
// ---------------------------------------------------------------------------
__global__ __launch_bounds__(256) void s4_scan_kernel(
    const float* __restrict__ z, const float* __restrict__ log_dt,
    const float* __restrict__ log_A_real, const float* __restrict__ A_imag,
    const float* __restrict__ C_re, const float* __restrict__ C_im,
    const float* __restrict__ Dskip, bf16_t* __restrict__ g) {
  int wv = (int)((blockIdx.x * blockDim.x + threadIdx.x) >> 5);  // (b,d) index
  int lane = threadIdx.x & 31;
  int b = wv / DD, d = wv % DD;

  float dt = __expf(log_dt[d]);
  int pi = d * NN + lane;
  float Are = -__expf(log_A_real[pi]);
  float Aim = A_imag[pi];
  float dtAre = dt * Are, dtAim = dt * Aim;
  float er = __expf(dtAre);
  float lr = er * __cosf(dtAim);
  float li = er * __sinf(dtAim);
  float wr = lr - 1.0f, wi = li;                    // exp(dtA)-1
  float den = Are * Are + Aim * Aim;                // |A|^2 (>0, Are<0)
  float qr = (wr * Are + wi * Aim) / den;           // (exp(dtA)-1)/A
  float qi = (wi * Are - wr * Aim) / den;
  float cre = C_re[pi], cim = C_im[pi];
  float cr = 2.0f * (cre * qr - cim * qi);          // fold the "2*" into Ctil
  float ci = 2.0f * (cre * qi + cim * qr);
  float ds = Dskip[d];

  const float* zrow = z + (size_t)wv * LL;
  bf16_t* grow = g + (size_t)b * LL * DD + d;

  float sre = 0.f, sim = 0.f;
  for (int l0 = 0; l0 < LL; l0 += 32) {
    float zv = zrow[l0 + lane];  // coalesced chunk; broadcast via shfl below
    float my = 0.f;
#pragma unroll 4
    for (int j = 0; j < 32; ++j) {
      float zj = __shfl(zv, j, 32);
      float nre = lr * sre - li * sim + zj;
      float nim = lr * sim + li * sre;
      sre = nre; sim = nim;
      float t = cr * sre - ci * sim;
#pragma unroll
      for (int off = 16; off > 0; off >>= 1) t += __shfl_xor(t, off, 32);
      if (lane == j) my = gelu_f(t + ds * zj);
    }
    grow[(size_t)(l0 + lane) * DD] = f2bf(my);
  }
}

// ---------------------------------------------------------------------------
// S4 output projection + GLU + residual. 16-position strip per workgroup,
// 8 waves; wave w computes the paired 16x16 tiles (e0=16w) and (e0+128) so
// the GLU combine a*sigmoid(g) happens in registers. K=128 -> 4 WMMA steps.
// ---------------------------------------------------------------------------
__global__ __launch_bounds__(256) void s4_glu_gemm(const bf16_t* __restrict__ g,
                                                   const bf16_t* __restrict__ wo,
                                                   const float* __restrict__ ob,
                                                   float* __restrict__ h) {
  int wv = threadIdx.x >> 5, lane = threadIdx.x & 31;
  int hi = lane >> 4, n = lane & 15;
  int b = blockIdx.x / (LL / 16);
  int l0 = (blockIdx.x % (LL / 16)) * 16;
  int e0 = wv * 16;

  const bf16_t* arow = g + ((size_t)b * LL + l0 + n) * DD;
  const bf16_t* wa = wo + (size_t)(e0 + n) * DD;         // value half
  const bf16_t* wg = wo + (size_t)(e0 + 128 + n) * DD;   // gate half
  v8f acc_a = {}; v8f acc_g = {};
#pragma unroll
  for (int k0 = 0; k0 < DD; k0 += 32) {
    v16bf af = load_a_frag(arow, k0, hi);
    v16bf ba = load_b_frag(wa, k0, hi);
    v16bf bg = load_b_frag(wg, k0, hi);
    acc_a = wmma_bf16(af, ba, acc_a);
    acc_g = wmma_bf16(af, bg, acc_g);
  }
  float bia = ob[e0 + n], big = ob[e0 + 128 + n];
  float* hp = h + ((size_t)b * DD + e0 + n) * LL + l0 + hi * 8;
  v4f h0 = *(v4f*)hp, h1 = *(v4f*)(hp + 4);
#pragma unroll
  for (int r = 0; r < 8; ++r) {
    float val = (acc_a[r] + bia) * sigmoid_f(acc_g[r] + big);
    if (r < 4) h0[r] += val; else h1[r - 4] += val;
  }
  *(v4f*)hp = h0; *(v4f*)(hp + 4) = h1;
}

// ---------------------------------------------------------------------------
// Fused FFN: t = gelu(W1 z + b1) staged as a bf16 16x256 tile in LDS, then
// h += W2 t + b2. Stage1: wave w -> tiles (e0, e0+128), K=128 (4 WMMA each).
// Stage2: wave w -> d-tile 16w, K=256 (8 WMMA) with A read from LDS.
// ---------------------------------------------------------------------------
__global__ __launch_bounds__(256) void ff_fused(const bf16_t* __restrict__ zb,
                                                const bf16_t* __restrict__ w1,
                                                const float* __restrict__ b1,
                                                const bf16_t* __restrict__ w2,
                                                const float* __restrict__ b2,
                                                float* __restrict__ h) {
  __shared__ bf16_t tls[16 * EE];  // 8 KB tile t[m][e]
  int wv = threadIdx.x >> 5, lane = threadIdx.x & 31;
  int hi = lane >> 4, n = lane & 15;
  int b = blockIdx.x / (LL / 16);
  int l0 = (blockIdx.x % (LL / 16)) * 16;
  int e0 = wv * 16;

  const bf16_t* arow = zb + ((size_t)b * LL + l0 + n) * DD;
  const bf16_t* wra = w1 + (size_t)(e0 + n) * DD;
  const bf16_t* wrb = w1 + (size_t)(e0 + 128 + n) * DD;
  v8f acc0 = {}; v8f acc1 = {};
#pragma unroll
  for (int k0 = 0; k0 < DD; k0 += 32) {
    v16bf af = load_a_frag(arow, k0, hi);
    v16bf bf0 = load_b_frag(wra, k0, hi);
    v16bf bf1 = load_b_frag(wrb, k0, hi);
    acc0 = wmma_bf16(af, bf0, acc0);
    acc1 = wmma_bf16(af, bf1, acc1);
  }
  float bb0 = b1[e0 + n], bb1 = b1[e0 + 128 + n];
#pragma unroll
  for (int r = 0; r < 8; ++r) {
    int m = hi * 8 + r;
    tls[m * EE + e0 + n]       = f2bf(gelu_f(acc0[r] + bb0));
    tls[m * EE + e0 + 128 + n] = f2bf(gelu_f(acc1[r] + bb1));
  }
  __syncthreads();

  int d0 = wv * 16;
  const bf16_t* w2row = w2 + (size_t)(d0 + n) * EE;
  const bf16_t* alds = &tls[n * EE];  // A row m = lane&15
  v8f acc = {};
#pragma unroll
  for (int k0 = 0; k0 < EE; k0 += 32) {
    v16bf af = load_a_frag(alds, k0, hi);
    v16bf bfr = load_b_frag(w2row, k0, hi);
    acc = wmma_bf16(af, bfr, acc);
  }
  float bb = b2[d0 + n];
  float* hp = h + ((size_t)b * DD + d0 + n) * LL + l0 + hi * 8;
  v4f h0 = *(v4f*)hp, h1 = *(v4f*)(hp + 4);
#pragma unroll
  for (int r = 0; r < 8; ++r) {
    float val = acc[r] + bb;
    if (r < 4) h0[r] += val; else h1[r - 4] += val;
  }
  *(v4f*)hp = h0; *(v4f*)(hp + 4) = h1;
}

// ---------------------------------------------------------------------------
// Head: logits = h[:,:,L-1] @ head_w.T + head_b, argmax -> int32[B]
// ---------------------------------------------------------------------------
__global__ __launch_bounds__(256) void head_kernel(const float* __restrict__ h,
                                                   const float* __restrict__ hw,
                                                   const float* __restrict__ hb,
                                                   int* __restrict__ out) {
  __shared__ float lg[BB][NCLS];
  int t = threadIdx.x;
  int b = t >> 4, c = t & 15;  // 256 threads == 16*16
  float s = hb[c];
  for (int d = 0; d < DD; ++d)
    s += h[((size_t)b * DD + d) * LL + (LL - 1)] * hw[c * DD + d];
  lg[b][c] = s;
  __syncthreads();
  if (t < BB) {
    int best = 0; float bv = lg[t][0];
    for (int cc = 1; cc < NCLS; ++cc)
      if (lg[t][cc] > bv) { bv = lg[t][cc]; best = cc; }
    out[t] = best;
  }
}

// ---------------------------------------------------------------------------
extern "C" void kernel_launch(void* const* d_in, const int* in_sizes, int n_in,
                              void* d_out, int out_size, void* d_ws, size_t ws_size,
                              hipStream_t stream) {
  const float* x             = (const float*)d_in[0];
  const float* conv_w        = (const float*)d_in[1];
  const float* conv_b        = (const float*)d_in[2];
  const float* s4_ln_w       = (const float*)d_in[3];
  const float* s4_ln_b       = (const float*)d_in[4];
  const float* s4_log_dt     = (const float*)d_in[5];
  const float* s4_log_A_real = (const float*)d_in[6];
  const float* s4_A_imag     = (const float*)d_in[7];
  const float* s4_C_re       = (const float*)d_in[8];
  const float* s4_C_im       = (const float*)d_in[9];
  const float* s4_Dp         = (const float*)d_in[10];
  const float* s4_out_w      = (const float*)d_in[11];
  const float* s4_out_b      = (const float*)d_in[12];
  const float* ff_ln_w       = (const float*)d_in[13];
  const float* ff_ln_b       = (const float*)d_in[14];
  const float* ff_w1         = (const float*)d_in[15];
  const float* ff_b1         = (const float*)d_in[16];
  const float* ff_w2         = (const float*)d_in[17];
  const float* ff_b2         = (const float*)d_in[18];
  const float* head_w        = (const float*)d_in[19];
  const float* head_b        = (const float*)d_in[20];
  int* out = (int*)d_out;

  // Workspace layout: h (64MB f32) | z (64MB f32) | g (32MB bf16) | bf16 weights
  const size_t nHDL = (size_t)BB * DD * LL;  // 16,777,216
  float* h   = (float*)d_ws;
  float* z   = h + nHDL;
  bf16_t* g  = (bf16_t*)(z + nHDL);          // (B,L,D) bf16 GEMM-A buffer
  bf16_t* wob = g + nHDL;
  const int WN = NBK * EE * DD;              // 196608 each
  bf16_t* w1b = wob + WN;
  bf16_t* w2b = w1b + WN;

  cvt_weights_kernel<<<(WN + 255) / 256, 256, 0, stream>>>(
      s4_out_w, ff_w1, ff_w2, wob, w1b, w2b, WN);
  conv_kernel<<<(int)(nHDL / 256), 256, 0, stream>>>(x, conv_w, conv_b, h);

  const int lnGrid = (BB * LL) / 256;        // 512
  const int gemmGrid = BB * (LL / 16);       // 8192
  const int scanGrid = (BB * DD) / 8;        // 256 blocks of 8 waves

  for (int i = 0; i < NBK; ++i) {
    ln_kernel<<<lnGrid, 256, 0, stream>>>(h, z, nullptr,
                                          s4_ln_w + i * DD, s4_ln_b + i * DD, 0);
    s4_scan_kernel<<<scanGrid, 256, 0, stream>>>(
        z, s4_log_dt + i * DD, s4_log_A_real + (size_t)i * DD * NN,
        s4_A_imag + (size_t)i * DD * NN, s4_C_re + (size_t)i * DD * NN,
        s4_C_im + (size_t)i * DD * NN, s4_Dp + i * DD, g);
    s4_glu_gemm<<<gemmGrid, 256, 0, stream>>>(
        g, wob + (size_t)i * EE * DD, s4_out_b + i * EE, h);
    ln_kernel<<<lnGrid, 256, 0, stream>>>(h, nullptr, g,
                                          ff_ln_w + i * DD, ff_ln_b + i * DD, 1);
    ff_fused<<<gemmGrid, 256, 0, stream>>>(
        g, w1b + (size_t)i * EE * DD, ff_b1 + i * EE,
        w2b + (size_t)i * DD * EE, ff_b2 + i * DD, h);
  }
  head_kernel<<<1, 256, 0, stream>>>(h, head_w, head_b, out);
}
